// STFT_60507499266734
// MI455X (gfx1250) — matrix operand order repaired and checked by reading, not compiled, not run on hardware
//
#include <hip/hip_runtime.h>

typedef __attribute__((ext_vector_type(16))) _Float16 v16h;
typedef __attribute__((ext_vector_type(8)))  _Float16 v8h;
typedef __attribute__((ext_vector_type(8)))  float    v8f;
typedef __attribute__((ext_vector_type(4)))  unsigned int u32x4;
typedef __attribute__((ext_vector_type(8)))  unsigned int u32x8;

#define N_FFT    2048
#define HOP      512
#define PAD_T    1024
#define T_LEN    262144
#define XP_LEN   (T_LEN + 2 * PAD_T)   // 264192 valid padded samples
#define XP_STRIDE 296448               // covers last-tile overrun (575*512+2047); 256-divisible
#define N_FRAMES 513                   // (262144 + 2*1024 - 2048)/512 + 1
#define N_BINS   1025                  // n_fft/2 + 1
#define N_OUT    2050                  // real rows then imag rows
#define N_PAD    2176                  // 17 * 128, padded basis rows (zeros past 2050)
#define BATCH    16
#define BK       64                    // K-chunk per TDM transfer
#define BM       96                    // M (frames) per block: 6 x 16 sub-tiles
#define M_SUB    6                     // BM / 16
#define M_TILES  6                     // ceil(513/96) -> 576 rows, same waste as BM=64
#define N_TILES  17                    // ceil(2050/128)

// ---------------- Kernel 1: windowed DFT basis, f16, k-major ----------------
__global__ void build_basis(const float* __restrict__ window,
                            _Float16* __restrict__ basis) {
  int idx = blockIdx.x * blockDim.x + threadIdx.x;
  if (idx >= N_PAD * N_FFT) return;
  int n = idx >> 11;            // / 2048
  int k = idx & (N_FFT - 1);
  const float C = 6.283185307179586f / (float)N_FFT;
  float v = 0.0f;
  float w = window[k];
  if (n < N_BINS) {
    int p = (n * k) & (N_FFT - 1);          // exact phase mod 2048
    v = w * __cosf((float)p * C);
  } else if (n < N_OUT) {
    int p = ((n - N_BINS) * k) & (N_FFT - 1);
    v = -w * __sinf((float)p * C);
  }
  basis[idx] = (_Float16)v;
}

// ------- Kernel 2: reflect-padded signal, f32 -> f16, tail-zeroed -----------
__global__ void build_xp(const float* __restrict__ x, _Float16* __restrict__ xp) {
  int e = blockIdx.x * blockDim.x + threadIdx.x;   // [0, XP_STRIDE)
  int b = blockIdx.y;
  float v = 0.0f;
  if (e < XP_LEN) {
    int jj = e - PAD_T;
    jj = (jj < 0) ? -jj : jj;                          // left reflect
    jj = (jj >= T_LEN) ? (2 * T_LEN - 2 - jj) : jj;    // right reflect
    v = x[(long)b * T_LEN + jj];
  }
  xp[(long)b * XP_STRIDE + e] = (_Float16)v;
}

// TDM 2D tile load: BM x BK f16 elements, row stride HOP, into LDS row-major.
// D# packing per CDNA5 ISA ch.8 (groups 2/3 zero -> tile_dim2/3 unused, 2D op).
__device__ __forceinline__ void tdm_load_2d(unsigned long long ga, unsigned int lds_off) {
  u32x4 g0;
  g0[0] = 1u;                                      // count=1 (valid), no gather
  g0[1] = lds_off;                                 // lds_addr (bytes)
  g0[2] = (unsigned int)ga;                        // global_addr[31:0]
  g0[3] = (unsigned int)(ga >> 32) | (2u << 30);   // global_addr[56:32] | type=2
  u32x8 g1;
  g1[0] = 1u << 16;                                // data_size=1 (2 bytes)
  g1[1] = 0u;                                      // abar=0 | tensor_dim0[15:0]=0
  g1[2] = 0x4000u;                                 // tensor_dim0=2^30 (hi16) | dim1 lo16
  g1[3] = 0x4000u | ((unsigned)BK << 16);          // tensor_dim1=2^30 (hi16) | tile_dim0=64
  g1[4] = (unsigned)BM;                            // tile_dim1=96 | tile_dim2=0
  g1[5] = (unsigned)HOP;                           // tensor_dim0_stride = 512 (lo32)
  g1[6] = 0u;                                      // stride0 hi16 | stride1 lo16
  g1[7] = 0u;                                      // stride1 hi32
  u32x4 gz = (u32x4){0u, 0u, 0u, 0u};
  asm volatile("tensor_load_to_lds %0, %1, %2, %3"
               :: "s"(g0), "s"(g1), "s"(gz), "s"(gz)
               : "memory");
}

__device__ __forceinline__ void wait_tensorcnt0() {
#if __has_builtin(__builtin_amdgcn_s_wait_tensorcnt)
  __builtin_amdgcn_s_wait_tensorcnt(0);
#else
  asm volatile("s_wait_tensorcnt 0x0" ::: "memory");
#endif
}

// ---- Kernel 3: WMMA GEMM, TDM double-buffered A, M-register-blocked --------
// Block = 8 waves. Tile M=96 frames x N=128 basis rows. Wave0 drives the TDM
// pipeline; per 32-K step each wave loads ONE B fragment (32B, L2-resident)
// and reuses it across 6 WMMAs.
__global__ __launch_bounds__(256) void stft_wmma(const _Float16* __restrict__ xp,
                                                 const _Float16* __restrict__ basis,
                                                 float* __restrict__ out) {
  __shared__ _Float16 As[2][BM][BK];   // double buffer: 2 x 12 KB

  const int bid = blockIdx.x;
  const int nt  = bid % N_TILES;
  const int mt  = (bid / N_TILES) % M_TILES;
  const int b   = bid / (N_TILES * M_TILES);

  const int tid  = threadIdx.x;
  const int lane = tid & 31;
  const int wave = tid >> 5;

  const int f0 = mt * BM;                            // first frame of tile
  const int n  = nt * 128 + wave * 16 + (lane & 15); // this lane's basis row

  // Fragment geometry (contiguous K order chosen here, A and B alike, wave32):
  // A: lane row = lane&15; lanes<16 hold K 0..15, lanes>=16 hold K 16..31
  // B: lane col = lane&15; lanes<16 hold K 0..15, lanes>=16 hold K 16..31
  const int m_frag = lane & 15;
  const int kgrp   = lane >> 4;     // 0 or 1
  const int a_bk   = kgrp * 16;
  const int b_bk   = kgrp * 16;

  const _Float16* bp_base = basis + (long)n * N_FFT;
  const unsigned long long ga_base =
      (unsigned long long)(size_t)(xp + (long)b * XP_STRIDE + (long)f0 * HOP);
  const unsigned int lds_base = (unsigned int)(size_t)&As[0][0][0];  // low 32b = LDS offset

  v8f acc[M_SUB];
  #pragma unroll
  for (int ms = 0; ms < M_SUB; ++ms) acc[ms] = (v8f){};

  const int NCHUNK = N_FFT / BK;   // 32
  if (wave == 0) tdm_load_2d(ga_base, lds_base);           // chunk 0 -> buf 0

  for (int c = 0; c < NCHUNK; ++c) {
    const int buf = c & 1;
    if (wave == 0) wait_tensorcnt0();                      // chunk c landed
    __syncthreads();                                       // data visible; buf^1 free
    if (wave == 0 && c + 1 < NCHUNK)                       // prefetch chunk c+1
      tdm_load_2d(ga_base + (unsigned long long)((c + 1) * BK) * 2u,
                  lds_base + (unsigned)(buf ^ 1) * (BM * BK * 2));

    const int kc = c * BK;
    #pragma unroll
    for (int kk = 0; kk < BK; kk += 32) {
      // One B fragment: contiguous 32B global read, reused by 6 WMMAs
      v16h bf = *(const v16h*)(bp_base + kc + kk + b_bk);
      #pragma unroll
      for (int ms = 0; ms < M_SUB; ++ms) {
        v8h alo = *(const v8h*)&As[buf][ms * 16 + m_frag][kk + a_bk];
        v8h ahi = *(const v8h*)&As[buf][ms * 16 + m_frag][kk + 8 + a_bk];
        v16h a;
        #pragma unroll
        for (int e = 0; e < 8; ++e) { a[e] = alo[e]; a[e + 8] = ahi[e]; }
        acc[ms] = __builtin_amdgcn_wmma_f32_16x16x32_f16(
            /*neg_a=*/false, a, /*neg_b=*/false, bf,
            /*c_mod=*/(short)0, acc[ms], /*reuse_a=*/false, /*reuse_b=*/false);
      }
    }
  }

  // Store D per C/D layout: VGPR r -> M = r (lanes<16) or r+8 (lanes>=16),
  // N = lane&15. Output: real block then imag block, each [B, 1025, 513].
  const long HALF = (long)BATCH * N_BINS * N_FRAMES;
  if (n < N_OUT) {
    long obase;
    if (n < N_BINS) obase = ((long)b * N_BINS + n) * N_FRAMES;
    else            obase = HALF + ((long)b * N_BINS + (n - N_BINS)) * N_FRAMES;
    #pragma unroll
    for (int ms = 0; ms < M_SUB; ++ms) {
      const int m_base = f0 + ms * 16 + kgrp * 8;
      #pragma unroll
      for (int r = 0; r < 8; ++r) {
        int m = m_base + r;
        if (m < N_FRAMES) out[obase + m] = acc[ms][r];
      }
    }
  }
}

extern "C" void kernel_launch(void* const* d_in, const int* in_sizes, int n_in,
                              void* d_out, int out_size, void* d_ws, size_t ws_size,
                              hipStream_t stream) {
  const float* x      = (const float*)d_in[0];   // [16, 262144] f32
  const float* window = (const float*)d_in[1];   // [2048] f32
  float*       out    = (float*)d_out;           // real||imag, [2,16,1025,513] f32

  // ws layout: basis f16 (8.9 MB) || xp f16 (9.5 MB)
  _Float16* basis = (_Float16*)d_ws;
  _Float16* xp    = basis + (long)N_PAD * N_FFT;

  int total = N_PAD * N_FFT;
  build_basis<<<(total + 255) / 256, 256, 0, stream>>>(window, basis);
  build_xp<<<dim3(XP_STRIDE / 256, BATCH), 256, 0, stream>>>(x, xp);

  dim3 grid(BATCH * M_TILES * N_TILES);          // 16*6*17 = 1632 blocks
  stft_wmma<<<grid, 256, 0, stream>>>(xp, basis, out);
}
